// LiquidTimeConstantLayer_87677462380890
// MI455X (gfx1250) — matrix operand order repaired and checked by reading, not compiled, not run on hardware
//
#include <hip/hip_runtime.h>
#include <math.h>

// ---------------- problem dims ----------------
constexpr int BB = 64;     // batch
constexpr int TT = 1024;   // time
constexpr int II = 128;    // input
constexpr int HH = 512;    // hidden
constexpr float DT = 0.1f;
constexpr float EPS = 1e-8f;
constexpr float BIG_CLAMP = 50.0f;

// ---------------- types ----------------
typedef __attribute__((ext_vector_type(16))) __bf16 v16bf;
typedef __attribute__((ext_vector_type(8)))  float  v8f;
typedef __attribute__((ext_vector_type(4)))  unsigned int v4u;

union Frag16 {
    v4u   u[2];
    v16bf v;
};

// f32 -> bf16 round-to-nearest-even
__device__ inline unsigned short f2bf(float x) {
    unsigned int u = __float_as_uint(x);
    unsigned int r = u + 0x7FFFu + ((u >> 16) & 1u);
    return (unsigned short)(r >> 16);
}

// Load A fragment: two 16B chunks at elem offsets k0 and k0+16 (bf16 elems)
__device__ inline v16bf load_frag_split(const unsigned short* base, int k0) {
    Frag16 f;
    f.u[0] = *(const v4u*)(base + k0);
    f.u[1] = *(const v4u*)(base + k0 + 16);
    return f.v;
}
// Load B fragment: 32B contiguous (16 bf16)
__device__ inline v16bf load_frag32(const unsigned short* p) {
    Frag16 f;
    f.u[0] = *(const v4u*)(p);
    f.u[1] = *(const v4u*)(p + 8);
    return f.v;
}

#define WMMA_BF16(a, b, c) \
    __builtin_amdgcn_wmma_f32_16x16x32_bf16(false, (a), false, (b), (short)0, (c), false, false)

// ---------------- workspace layout ----------------
constexpr size_t alignup(size_t x) { return (x + 255) & ~size_t(255); }
constexpr size_t OFF_BAR  = 0;                                         // barrier counter
constexpr size_t OFF_HB   = 256;                                       // h bf16, double buffered: 2*B*H
constexpr size_t OFF_WH   = OFF_HB  + alignup((size_t)2 * BB * HH * 2);// 4 x (H,H) bf16: iw, im, is, tch
constexpr size_t OFF_WX   = OFF_WH  + alignup((size_t)4 * HH * HH * 2);// 4 x (H,I) bf16: sw, sm, ss, tcx
constexpr size_t OFF_XB   = OFF_WX  + alignup((size_t)4 * HH * II * 2);// x bf16 (B*T, I)
constexpr size_t OFF_SENS = OFF_XB  + alignup((size_t)BB * TT * II * 2);// sensory f32 (T,B,H)
constexpr size_t OFF_TCX  = OFF_SENS + alignup((size_t)TT * BB * HH * 4);// tcx f32 (T,B,H)

// ---------------- kernels ----------------

__global__ __launch_bounds__(256) void init_kernel(unsigned short* hb, unsigned int* bar) {
    int i = blockIdx.x * blockDim.x + threadIdx.x;
    if (i < 2 * BB * HH) hb[i] = 0x3F80;  // bf16(1.0) for both ping-pong buffers
    if (i == 0) *bar = 0u;
}

__global__ __launch_bounds__(256) void cvt_kernel(const float* __restrict__ src,
                                                  unsigned short* __restrict__ dst, int n) {
    int i = blockIdx.x * blockDim.x + threadIdx.x;
    if (i < n) dst[i] = f2bf(src[i]);
}

// Phase 1: x-projections.  A block owns one 16-column tile; its 4 weight slices
// (4 x 16 x 128 bf16 = 16 KB) are staged in LDS once.  Each wave computes two 16-row
// tiles so each LDS B-fragment feeds 2 WMMAs and each A-fragment feeds 4 weights.
// Writes sens and tcx in (T, B, H) layout for the scan.
__global__ __launch_bounds__(256) void xproj_kernel(
    const unsigned short* __restrict__ xb,    // (B*T, I) bf16
    const unsigned short* __restrict__ wx,    // 4 x (H, I) bf16
    const float* __restrict__ bsw, const float* __restrict__ bsm,
    const float* __restrict__ bss, const float* __restrict__ btc,
    float* __restrict__ sens, float* __restrict__ tcx) {
    __shared__ __align__(16) unsigned short Wxl[4 * 16 * II];  // [w][n_local][k] = 16 KB

    const int ctile = blockIdx.x & 31;        // H/16 = 32 column tiles
    const int rgrp  = blockIdx.x >> 5;        // 256 row groups of 256 rows
    const int n0  = ctile * 16;
    const int tid = threadIdx.x;

    for (int i = tid; i < 4 * 16 * II; i += 256) {
        int w   = i / (16 * II);
        int rem = i - w * 16 * II;
        int n   = rem / II;
        int k   = rem - n * II;
        Wxl[i] = wx[(size_t)w * HH * II + (size_t)(n0 + n) * II + k];
    }
    __syncthreads();

    const int lane = tid & 31;
    const int wave = tid >> 5;
    const int hi = lane >> 4;
    const int l  = lane & 15;
    const int R0 = rgrp * 256 + wave * 32;    // this wave: rows R0..R0+31 (two tiles)

    const unsigned short* arow0 = xb + (size_t)(R0 + l) * II;
    const unsigned short* arow1 = xb + (size_t)(R0 + 16 + l) * II;
    const unsigned short* wl0 = &Wxl[(0 * 16 + l) * II];
    const unsigned short* wl1 = &Wxl[(1 * 16 + l) * II];
    const unsigned short* wl2 = &Wxl[(2 * 16 + l) * II];
    const unsigned short* wl3 = &Wxl[(3 * 16 + l) * II];

    v8f a0w = {}, a0m = {}, a0s = {}, a0t = {};
    v8f a1w = {}, a1m = {}, a1s = {}, a1t = {};
#pragma unroll
    for (int ki = 0; ki < II / 32; ++ki) {
        v16bf a0 = load_frag_split(arow0, ki * 32 + hi * 8);
        v16bf a1 = load_frag_split(arow1, ki * 32 + hi * 8);
        int kb = ki * 32 + hi * 8;
        v16bf b0 = load_frag_split(wl0, kb);
        v16bf b1 = load_frag_split(wl1, kb);
        v16bf b2 = load_frag_split(wl2, kb);
        v16bf b3 = load_frag_split(wl3, kb);
        a0w = WMMA_BF16(a0, b0, a0w);  a1w = WMMA_BF16(a1, b0, a1w);
        a0m = WMMA_BF16(a0, b1, a0m);  a1m = WMMA_BF16(a1, b1, a1m);
        a0s = WMMA_BF16(a0, b2, a0s);  a1s = WMMA_BF16(a1, b2, a1s);
        a0t = WMMA_BF16(a0, b3, a0t);  a1t = WMMA_BF16(a1, b3, a1t);
    }

    const int b   = R0 / TT;     // T % 32 == 0 -> both tiles share the batch index
    const int t00 = R0 % TT;
    const int col = n0 + l;
    const float vbw = bsw[col], vbm = bsm[col], vbs = bss[col], vbt = btc[col];
#pragma unroll
    for (int r = 0; r < 2; ++r) {
        const v8f& cw = r ? a1w : a0w;
        const v8f& cm = r ? a1m : a0m;
        const v8f& cs = r ? a1s : a0s;
        const v8f& ct = r ? a1t : a0t;
#pragma unroll
        for (int v = 0; v < 8; ++v) {
            int t = t00 + r * 16 + v + hi * 8;
            size_t addr = ((size_t)t * BB + b) * HH + col;
            float sw = cw[v] + vbw;
            float sm = cm[v] + vbm;
            float ss = cs[v] + vbs;
            float sig = 1.0f / (1.0f + __expf(-sm));
            float se  = __expf(fminf(ss, BIG_CLAMP));
            sens[addr] = sw * sig * se;
            tcx[addr]  = ct[v] + vbt;
        }
    }
}

// Phase 2: sequential scan.  32 blocks x 128 threads.  Block owns 16 H-columns; its 4
// weight slices (64 KB bf16) live in LDS for all 1024 steps.  h state lives in registers
// (fragment layout), shared cross-block via double-buffered bf16 h + grid barrier.
// sens/tcx are prefetched before the K-loop and the A-fragment load is software-
// pipelined so global latency overlaps the 64 WMMAs on the serial critical path.
__global__ __launch_bounds__(128) void scan_kernel(
    const unsigned short* __restrict__ wh,    // 4 x (H,H) bf16: iw, im, is, tch
    const float* __restrict__ biw, const float* __restrict__ bim, const float* __restrict__ bis,
    const float* __restrict__ sens, const float* __restrict__ tcx,
    unsigned short* __restrict__ hb,          // 2 x (B,H) bf16 ping-pong
    float* __restrict__ out,                  // (B,T,H)
    unsigned int* __restrict__ bar) {
    __shared__ __align__(16) unsigned short Wloc[4 * 16 * HH];  // [w][n_local][k] = 64 KB

    const int n0  = blockIdx.x * 16;
    const int tid = threadIdx.x;
    for (int i = tid; i < 4 * 16 * HH; i += 128) {
        int w   = i / (16 * HH);
        int rem = i - w * 16 * HH;
        int n   = rem / HH;
        int k   = rem - n * HH;
        Wloc[i] = wh[(size_t)w * HH * HH + (size_t)(n0 + n) * HH + k];
    }
    __syncthreads();

    const int lane = tid & 31;
    const int wave = tid >> 5;    // row tile: rows wave*16 .. wave*16+15
    const int hi = lane >> 4;
    const int l  = lane & 15;
    const int col = n0 + l;
    const float vbw = biw[col], vbm = bim[col], vbs = bis[col];

    const unsigned short* wl0 = &Wloc[(0 * 16 + l) * HH];
    const unsigned short* wl1 = &Wloc[(1 * 16 + l) * HH];
    const unsigned short* wl2 = &Wloc[(2 * 16 + l) * HH];
    const unsigned short* wl3 = &Wloc[(3 * 16 + l) * HH];

    // h state for this lane's 8 fragment elements: row = wave*16 + v + hi*8, col = col
    float hloc[8];
#pragma unroll
    for (int v = 0; v < 8; ++v) hloc[v] = 1.0f;

    const unsigned int nb = gridDim.x;

    for (int t = 0; t < TT; ++t) {
        const unsigned short* hcur = hb + (size_t)(t & 1) * BB * HH;
        unsigned short* hnxt       = hb + (size_t)((t + 1) & 1) * BB * HH;
        const unsigned short* arow = hcur + (size_t)(wave * 16 + l) * HH;

        // Prefetch epilogue operands so they overlap the WMMA K-loop.
        const size_t tb = (size_t)t * BB;
        float s_pre[8], tc_pre[8];
#pragma unroll
        for (int v = 0; v < 8; ++v) {
            int brow = wave * 16 + v + hi * 8;
            size_t sidx = (tb + brow) * HH + col;
            s_pre[v]  = sens[sidx];
            tc_pre[v] = tcx[sidx];
        }

        v8f aw = {}, am = {}, as_ = {}, at = {};
        v16bf a = load_frag_split(arow, hi * 8);   // ki = 0 fragment
#pragma unroll
        for (int ki = 0; ki < HH / 32; ++ki) {
            // software pipeline: issue next A load before this iteration's WMMAs
            v16bf a_next = (ki + 1 < HH / 32)
                               ? load_frag_split(arow, (ki + 1) * 32 + hi * 8)
                               : a;
            int kb = ki * 32 + hi * 8;
            v16bf b0 = load_frag_split(wl0, kb);
            v16bf b1 = load_frag_split(wl1, kb);
            v16bf b2 = load_frag_split(wl2, kb);
            v16bf b3 = load_frag_split(wl3, kb);
            aw  = WMMA_BF16(a, b0, aw);
            am  = WMMA_BF16(a, b1, am);
            as_ = WMMA_BF16(a, b2, as_);
            at  = WMMA_BF16(a, b3, at);
            a = a_next;
        }

#pragma unroll
        for (int v = 0; v < 8; ++v) {
            int brow = wave * 16 + v + hi * 8;
            float hold = hloc[v];
            float ti = tc_pre[v] + at[v];                 // tcx_t + h @ Wtch.T
            float sp = (ti > 20.0f) ? ti : log1pf(__expf(ti));
            float tau = sp + 0.1f;
            float sig = 1.0f / (1.0f + __expf(-(am[v] + vbm)));
            float se  = __expf(fminf(as_[v] + vbs, BIG_CLAMP));
            float inter = (aw[v] + vbw) * sig * se;
            float dh = (s_pre[v] + inter - hold) / fmaxf(tau, EPS);
            float hnew = hold + DT * dh;
            hloc[v] = hnew;
            hnxt[(size_t)brow * HH + col] = f2bf(hnew);
            out[((size_t)brow * TT + t) * HH + col] = hnew;
        }

        // grid-wide barrier (monotonic counter; double-buffered h needs only one per step)
        __syncthreads();
        if (tid == 0) {
            __builtin_amdgcn_fence(__ATOMIC_RELEASE, "agent");
            __hip_atomic_fetch_add(bar, 1u, __ATOMIC_RELAXED, __HIP_MEMORY_SCOPE_AGENT);
            unsigned int target = (unsigned int)(t + 1) * nb;
            while (__hip_atomic_load(bar, __ATOMIC_RELAXED, __HIP_MEMORY_SCOPE_AGENT) < target)
                __builtin_amdgcn_s_sleep(1);
            __builtin_amdgcn_fence(__ATOMIC_ACQUIRE, "agent");
        }
        __syncthreads();
    }
}

// ---------------- launch ----------------
extern "C" void kernel_launch(void* const* d_in, const int* in_sizes, int n_in,
                              void* d_out, int out_size, void* d_ws, size_t ws_size,
                              hipStream_t stream) {
    (void)in_sizes; (void)n_in; (void)out_size; (void)ws_size;
    const float* x    = (const float*)d_in[0];
    const float* Wsw  = (const float*)d_in[1];
    const float* bsw  = (const float*)d_in[2];
    const float* Wsm  = (const float*)d_in[3];
    const float* bsm  = (const float*)d_in[4];
    const float* Wss  = (const float*)d_in[5];
    const float* bss  = (const float*)d_in[6];
    const float* Wiw  = (const float*)d_in[7];
    const float* biw  = (const float*)d_in[8];
    const float* Wim  = (const float*)d_in[9];
    const float* bim  = (const float*)d_in[10];
    const float* Wis  = (const float*)d_in[11];
    const float* bis  = (const float*)d_in[12];
    const float* Wtcx = (const float*)d_in[13];
    const float* Wtch = (const float*)d_in[14];
    const float* btc  = (const float*)d_in[15];

    char* ws = (char*)d_ws;
    unsigned int*   bar  = (unsigned int*)(ws + OFF_BAR);
    unsigned short* hb   = (unsigned short*)(ws + OFF_HB);
    unsigned short* wh   = (unsigned short*)(ws + OFF_WH);
    unsigned short* wxb  = (unsigned short*)(ws + OFF_WX);
    unsigned short* xb   = (unsigned short*)(ws + OFF_XB);
    float*          sens = (float*)(ws + OFF_SENS);
    float*          tcxb = (float*)(ws + OFF_TCX);

    init_kernel<<<(2 * BB * HH + 255) / 256, 256, 0, stream>>>(hb, bar);

    const int nHH = HH * HH, nHI = HH * II, nX = BB * TT * II;
    cvt_kernel<<<(nHH + 255) / 256, 256, 0, stream>>>(Wiw,  wh + 0 * (size_t)nHH, nHH);
    cvt_kernel<<<(nHH + 255) / 256, 256, 0, stream>>>(Wim,  wh + 1 * (size_t)nHH, nHH);
    cvt_kernel<<<(nHH + 255) / 256, 256, 0, stream>>>(Wis,  wh + 2 * (size_t)nHH, nHH);
    cvt_kernel<<<(nHH + 255) / 256, 256, 0, stream>>>(Wtch, wh + 3 * (size_t)nHH, nHH);
    cvt_kernel<<<(nHI + 255) / 256, 256, 0, stream>>>(Wsw,  wxb + 0 * (size_t)nHI, nHI);
    cvt_kernel<<<(nHI + 255) / 256, 256, 0, stream>>>(Wsm,  wxb + 1 * (size_t)nHI, nHI);
    cvt_kernel<<<(nHI + 255) / 256, 256, 0, stream>>>(Wss,  wxb + 2 * (size_t)nHI, nHI);
    cvt_kernel<<<(nHI + 255) / 256, 256, 0, stream>>>(Wtcx, wxb + 3 * (size_t)nHI, nHI);
    cvt_kernel<<<(nX + 255) / 256, 256, 0, stream>>>(x, xb, nX);

    // 32 column tiles x 256 row groups (256 rows each, 2 row tiles per wave)
    xproj_kernel<<<32 * (BB * TT / 256), 256, 0, stream>>>(
        xb, wxb, bsw, bsm, bss, btc, sens, tcxb);

    scan_kernel<<<HH / 16, 128, 0, stream>>>(
        wh, biw, bim, bis, sens, tcxb, hb, (float*)d_out, bar);
}
